// CausalSelfAttention_14456859918921
// MI455X (gfx1250) — matrix operand order repaired and checked by reading, not compiled, not run on hardware
//
#include <hip/hip_runtime.h>
#include <hip/hip_bf16.h>

// ---------------- problem constants ----------------
#define BATCH 4
#define SEQ   2048
#define CH    1024
#define NH    16
#define HD    64
#define MROWS (BATCH * SEQ)      // 8192
#define N_QKV (3 * CH)           // 3072

typedef __bf16 bf16_t;
typedef __attribute__((ext_vector_type(16))) __bf16 v16bf;
typedef __attribute__((ext_vector_type(8)))  __bf16 v8bf;
typedef __attribute__((ext_vector_type(8)))  float  v8f;

__device__ __forceinline__ v16bf join16(v8bf lo, v8bf hi) {
    return __builtin_shufflevector(lo, hi, 0, 1, 2, 3, 4, 5, 6, 7,
                                           8, 9, 10, 11, 12, 13, 14, 15);
}

__device__ __forceinline__ v8f wmma_bf16(v16bf a, v16bf b, v8f c) {
    return __builtin_amdgcn_wmma_f32_16x16x32_bf16(
        /*neg_a=*/false, a, /*neg_b=*/false, b,
        /*c_mod=*/(short)0, c, /*reuse_a=*/false, /*reuse_b=*/false);
}

// A-matrix operand (16x32 bf16): lane holds row (lane&15); k-chunks at
// k0 + 8*(lane>>4) and k0 + 16 + 8*(lane>>4).
__device__ __forceinline__ v16bf load_a16x32(const bf16_t* row, int k0, int hl) {
    v8bf lo = *(const v8bf*)(row + k0 + 8 * hl);
    v8bf hi = *(const v8bf*)(row + k0 + 16 + 8 * hl);
    return join16(lo, hi);
}

// B-matrix operand (32x16 bf16): lane holds column (lane&15); k-chunks at
// k0 + 8*(lane>>4) and k0 + 16 + 8*(lane>>4).
__device__ __forceinline__ v16bf load_b32x16(const bf16_t* col, int k0, int hl) {
    v8bf lo = *(const v8bf*)(col + k0 + 8 * hl);
    v8bf hi = *(const v8bf*)(col + k0 + 16 + 8 * hl);
    return join16(lo, hi);
}

// ---------------- conversion kernels ----------------
__global__ void cvt_f32_bf16(const float* __restrict__ in,
                             bf16_t* __restrict__ out, int n) {
    int i = blockIdx.x * blockDim.x + threadIdx.x;
    if (i < n) out[i] = (bf16_t)in[i];
}

// in: [K][N] f32 row-major  ->  out: [N][K] bf16 row-major
__global__ void cvt_transpose_bf16(const float* __restrict__ in,
                                   bf16_t* __restrict__ out, int K, int N) {
    int i = blockIdx.x * blockDim.x + threadIdx.x;
    if (i < K * N) {
        int n = i / K, k = i % K;
        out[i] = (bf16_t)in[(size_t)k * N + n];
    }
}

// ---------------- fused QKV projection GEMM (64x32 tile per wave) --------
// A: x_bf16 [MROWS][CH], Bt: Wqkv^T [N_QKV][CH], bias f32 [N_QKV]
// Scatters into Q[B,H,T,D] (pre-scaled by 1/sqrt(D)), K[B,H,T,D], Vt[B,H,D,T]
__global__ void __launch_bounds__(128, 2)
gemm_qkv(const bf16_t* __restrict__ A,
         const bf16_t* __restrict__ Bt,
         const float* __restrict__ bias,
         bf16_t* __restrict__ Qb,
         bf16_t* __restrict__ Kb,
         bf16_t* __restrict__ Vt) {
    const int lane = threadIdx.x & 31;
    const int wave = threadIdx.x >> 5;
    const int ln = lane & 15, hl = lane >> 4;
    const int n0 = (blockIdx.x * 4 + wave) * 32;
    const int m0 = blockIdx.y * 64;

    const bf16_t* arow[4];
    const bf16_t* brow[2];
#pragma unroll
    for (int mi = 0; mi < 4; ++mi)
        arow[mi] = A + (size_t)(m0 + mi * 16 + ln) * CH;
#pragma unroll
    for (int ni = 0; ni < 2; ++ni)
        brow[ni] = Bt + (size_t)(n0 + ni * 16 + ln) * CH;

    v8f acc[4][2];
#pragma unroll
    for (int mi = 0; mi < 4; ++mi)
#pragma unroll
        for (int ni = 0; ni < 2; ++ni) acc[mi][ni] = (v8f){};

#pragma unroll 2
    for (int k0 = 0; k0 < CH; k0 += 32) {
        v16bf av[4], bv[2];
#pragma unroll
        for (int mi = 0; mi < 4; ++mi) av[mi] = load_a16x32(arow[mi], k0, hl);
#pragma unroll
        for (int ni = 0; ni < 2; ++ni) bv[ni] = load_b32x16(brow[ni], k0, hl);
#pragma unroll
        for (int mi = 0; mi < 4; ++mi)
#pragma unroll
            for (int ni = 0; ni < 2; ++ni)
                acc[mi][ni] = wmma_bf16(av[mi], bv[ni], acc[mi][ni]);
    }

    // ---- epilogue: all tile-routing values are uniform per 16-wide tile ----
    const int b = m0 >> 11;              // scalar: 64-row tile never crosses batch
    const int tbase = m0 & (SEQ - 1);    // scalar
#pragma unroll
    for (int ni = 0; ni < 2; ++ni) {
        const int jbase = n0 + ni * 16;          // scalar
        const int three = jbase >> 10;           // scalar: 0=q 1=k 2=v
        const int h = (jbase >> 6) & (NH - 1);   // scalar
        const int dbase = jbase & (HD - 1);      // scalar; d = dbase + ln
        const float bj = bias[jbase + ln];
        const size_t bh = (size_t)b * NH + h;

        if (three == 0) {
            bf16_t* dst = Qb + (bh * SEQ + tbase) * HD + dbase + ln;
#pragma unroll
            for (int mi = 0; mi < 4; ++mi)
#pragma unroll
                for (int i = 0; i < 8; ++i) {
                    const int t = mi * 16 + i + 8 * hl;
                    dst[(size_t)t * HD] = (bf16_t)((acc[mi][ni][i] + bj) * 0.125f);
                }
        } else if (three == 1) {
            bf16_t* dst = Kb + (bh * SEQ + tbase) * HD + dbase + ln;
#pragma unroll
            for (int mi = 0; mi < 4; ++mi)
#pragma unroll
                for (int i = 0; i < 8; ++i) {
                    const int t = mi * 16 + i + 8 * hl;
                    dst[(size_t)t * HD] = (bf16_t)(acc[mi][ni][i] + bj);
                }
        } else {
            bf16_t* dst = Vt + (bh * HD + dbase + ln) * SEQ + tbase;
#pragma unroll
            for (int mi = 0; mi < 4; ++mi)
#pragma unroll
                for (int i = 0; i < 8; ++i) {
                    const int t = mi * 16 + i + 8 * hl;
                    dst[t] = (bf16_t)(acc[mi][ni][i] + bj);
                }
        }
    }
}

// ---------------- flash attention (one wave per 16-row query tile) --------
// Qb,Kb: [BH][T][D] bf16 (Q pre-scaled), Vt: [BH][D][T] bf16
// out: attn [B][T][C] bf16
__global__ void attn_flash(const bf16_t* __restrict__ Qb,
                           const bf16_t* __restrict__ Kb,
                           const bf16_t* __restrict__ Vt,
                           bf16_t* __restrict__ attnb) {
    __shared__ bf16_t pb_all[4][16 * 32];   // per-wave P transpose staging

    const int lane = threadIdx.x & 31;
    const int wave = threadIdx.x >> 5;
    const int ln = lane & 15, hl = lane >> 4;
    bf16_t* pb = pb_all[wave];

    const int bh = blockIdx.y;              // 0..63
    const int b = bh >> 4, h = bh & (NH - 1);
    const int q0 = (blockIdx.x * 4 + wave) * 16;

    // Q as two A-matrix operands (d 0..31 and 32..63), loaded once
    const bf16_t* qrow = Qb + ((size_t)bh * SEQ + q0 + ln) * HD;
    v16bf qa[2];
#pragma unroll
    for (int dh = 0; dh < 2; ++dh) qa[dh] = load_a16x32(qrow, 32 * dh, hl);

    v8f o[4];                               // O tile: 16 rows x 64 dims
#pragma unroll
    for (int nt = 0; nt < 4; ++nt) o[nt] = (v8f){};
    float mrow[8], lrow[8];
#pragma unroll
    for (int i = 0; i < 8; ++i) { mrow[i] = -INFINITY; lrow[i] = 0.f; }

    for (int kb = 0; kb < q0 + 16; kb += 32) {
        // ---- S = Q * K^T for two 16-key column tiles ----
        v8f s[2];
#pragma unroll
        for (int t = 0; t < 2; ++t) {
            const bf16_t* krow = Kb + ((size_t)bh * SEQ + kb + 16 * t + ln) * HD;
            v8f sacc = {};
#pragma unroll
            for (int dh = 0; dh < 2; ++dh) {
                v16bf kv = load_b32x16(krow, 32 * dh, hl);
                sacc = wmma_bf16(qa[dh], kv, sacc);
            }
            // causal mask: this lane holds key column (kb + 16t + ln)
            const int key = kb + 16 * t + ln;
#pragma unroll
            for (int i = 0; i < 8; ++i) {
                const int row = q0 + i + 8 * hl;
                if (key > row) sacc[i] = -INFINITY;
            }
            s[t] = sacc;
        }

        // ---- online softmax (row = VGPR i + 8*hl; reduce across 16 lanes) ----
#pragma unroll
        for (int i = 0; i < 8; ++i) {
            float rmax = fmaxf(s[0][i], s[1][i]);
#pragma unroll
            for (int m = 1; m < 16; m <<= 1)
                rmax = fmaxf(rmax, __shfl_xor(rmax, m, 32));
            const float mnew = fmaxf(mrow[i], rmax);
            const float sc = __expf(mrow[i] - mnew);
            const float p0 = __expf(s[0][i] - mnew);
            const float p1 = __expf(s[1][i] - mnew);
            float rsum = p0 + p1;
#pragma unroll
            for (int m = 1; m < 16; m <<= 1)
                rsum += __shfl_xor(rsum, m, 32);
            lrow[i] = lrow[i] * sc + rsum;
            mrow[i] = mnew;
            s[0][i] = p0;
            s[1][i] = p1;
#pragma unroll
            for (int nt = 0; nt < 4; ++nt) o[nt][i] *= sc;
        }

        // ---- transpose P (C-layout -> A-layout) through per-wave LDS ----
        // ensure previous iteration's LDS reads have completed
        asm volatile("s_wait_dscnt 0" ::: "memory");
#pragma unroll
        for (int t = 0; t < 2; ++t)
#pragma unroll
            for (int i = 0; i < 8; ++i)
                pb[(i + 8 * hl) * 32 + 16 * t + ln] = (bf16_t)s[t][i];
        asm volatile("s_wait_dscnt 0" ::: "memory");

        v8bf plo = *(const v8bf*)(pb + ln * 32 + 8 * hl);
        v8bf phi = *(const v8bf*)(pb + ln * 32 + 16 + 8 * hl);
        v16bf pa = join16(plo, phi);

        // ---- O += P * V  (V read transposed: contiguous in key dim) ----
#pragma unroll
        for (int nt = 0; nt < 4; ++nt) {
            const bf16_t* vrow =
                Vt + ((size_t)bh * HD + nt * 16 + ln) * SEQ + kb;
            v16bf vv = load_b32x16(vrow, 0, hl);
            o[nt] = wmma_bf16(pa, vv, o[nt]);
        }
    }

    // ---- epilogue: normalize and scatter to [B][T][C] ----
    float inv[8];
#pragma unroll
    for (int i = 0; i < 8; ++i) inv[i] = 1.0f / lrow[i];
#pragma unroll
    for (int nt = 0; nt < 4; ++nt) {
        bf16_t* dst = attnb + ((size_t)b * SEQ + q0) * CH + h * HD + nt * 16 + ln;
#pragma unroll
        for (int i = 0; i < 8; ++i) {
            const int r = i + 8 * hl;
            dst[(size_t)r * CH] = (bf16_t)(o[nt][i] * inv[i]);
        }
    }
}

// ---------------- output projection GEMM (64x32 tile per wave) -----------
// A: attn bf16 [MROWS][CH], Bt: Wproj^T [CH][CH], bias f32 [CH] -> out f32
__global__ void __launch_bounds__(128, 2)
gemm_proj(const bf16_t* __restrict__ A,
          const bf16_t* __restrict__ Bt,
          const float* __restrict__ bias,
          float* __restrict__ out) {
    const int lane = threadIdx.x & 31;
    const int wave = threadIdx.x >> 5;
    const int ln = lane & 15, hl = lane >> 4;
    const int n0 = (blockIdx.x * 4 + wave) * 32;
    const int m0 = blockIdx.y * 64;

    const bf16_t* arow[4];
    const bf16_t* brow[2];
#pragma unroll
    for (int mi = 0; mi < 4; ++mi)
        arow[mi] = A + (size_t)(m0 + mi * 16 + ln) * CH;
#pragma unroll
    for (int ni = 0; ni < 2; ++ni)
        brow[ni] = Bt + (size_t)(n0 + ni * 16 + ln) * CH;

    v8f acc[4][2];
#pragma unroll
    for (int mi = 0; mi < 4; ++mi)
#pragma unroll
        for (int ni = 0; ni < 2; ++ni) acc[mi][ni] = (v8f){};

#pragma unroll 2
    for (int k0 = 0; k0 < CH; k0 += 32) {
        v16bf av[4], bv[2];
#pragma unroll
        for (int mi = 0; mi < 4; ++mi) av[mi] = load_a16x32(arow[mi], k0, hl);
#pragma unroll
        for (int ni = 0; ni < 2; ++ni) bv[ni] = load_b32x16(brow[ni], k0, hl);
#pragma unroll
        for (int mi = 0; mi < 4; ++mi)
#pragma unroll
            for (int ni = 0; ni < 2; ++ni)
                acc[mi][ni] = wmma_bf16(av[mi], bv[ni], acc[mi][ni]);
    }

#pragma unroll
    for (int ni = 0; ni < 2; ++ni) {
        const int j = n0 + ni * 16 + ln;
        const float bj = bias[j];
        float* dst = out + (size_t)m0 * CH + j;
#pragma unroll
        for (int mi = 0; mi < 4; ++mi)
#pragma unroll
            for (int i = 0; i < 8; ++i) {
                const int r = mi * 16 + i + 8 * hl;
                dst[(size_t)r * CH] = acc[mi][ni][i] + bj;
            }
    }
}

// ---------------- host launcher ----------------
extern "C" void kernel_launch(void* const* d_in, const int* in_sizes, int n_in,
                              void* d_out, int out_size, void* d_ws, size_t ws_size,
                              hipStream_t stream) {
    const float* x     = (const float*)d_in[0];   // [B,T,C]
    const float* Wqkv  = (const float*)d_in[1];   // [C,3C]
    const float* bqkv  = (const float*)d_in[2];   // [3C]
    const float* Wproj = (const float*)d_in[3];   // [C,C]
    const float* bproj = (const float*)d_in[4];   // [C]
    float* out = (float*)d_out;                   // [B,T,C]

    size_t off = 0;
    auto carve = [&](size_t elems) -> bf16_t* {
        bf16_t* p = (bf16_t*)((char*)d_ws + off);
        off = (off + elems * sizeof(bf16_t) + 255) & ~(size_t)255;
        return p;
    };
    const size_t nX   = (size_t)MROWS * CH;              // 8.4M
    const size_t nWq  = (size_t)N_QKV * CH;              // 3.1M
    const size_t nWp  = (size_t)CH * CH;                 // 1.0M
    const size_t nHead = (size_t)BATCH * NH * SEQ * HD;  // 8.4M

    bf16_t* xb     = carve(nX);
    bf16_t* wqkvt  = carve(nWq);
    bf16_t* wprojt = carve(nWp);
    bf16_t* Qb     = carve(nHead);
    bf16_t* Kb     = carve(nHead);
    bf16_t* Vt     = carve(nHead);
    bf16_t* attnb  = xb;   // reuse: x_bf16 dead after gemm_qkv

    // 1) conversions
    cvt_f32_bf16<<<dim3((nX + 255) / 256), dim3(256), 0, stream>>>(x, xb, (int)nX);
    cvt_transpose_bf16<<<dim3((nWq + 255) / 256), dim3(256), 0, stream>>>(
        Wqkv, wqkvt, CH, N_QKV);
    cvt_transpose_bf16<<<dim3((nWp + 255) / 256), dim3(256), 0, stream>>>(
        Wproj, wprojt, CH, CH);

    // 2) fused QKV projection: M=8192, N=3072, 64x32 tile per wave
    gemm_qkv<<<dim3(N_QKV / 32 / 4, MROWS / 64), dim3(128), 0, stream>>>(
        xb, wqkvt, bqkv, Qb, Kb, Vt);

    // 3) flash attention: 128 query tiles per (b,h), 4 waves/block
    attn_flash<<<dim3(SEQ / 16 / 4, BATCH * NH), dim3(128), 0, stream>>>(
        Qb, Kb, Vt, attnb);

    // 4) output projection: M=8192, N=1024, 64x32 tile per wave
    gemm_proj<<<dim3(CH / 32 / 4, MROWS / 64), dim3(128), 0, stream>>>(
        attnb, wprojt, bproj, out);
}
